// PrefetchHyperbolicAttention_16870631539368
// MI455X (gfx1250) — matrix operand order repaired and checked by reading, not compiled, not run on hardware
//
#include <hip/hip_runtime.h>
#include <hip/hip_bf16.h>

// ---------------------------------------------------------------------------
// Hyperbolic attention, MI455X (gfx1250, wave32, WMMA).
// B=1, N=2048, D=1024, H=16, hd=64, curvature=1.
// All matmuls: v_wmma_f32_16x16x32_f16 (f16 in, f32 accum).
// WMMA fragments loaded as 16-byte pairs (ds_load_b128 from LDS; Q from global).
// Tile staging: double-buffered GLOBAL_LOAD_ASYNC_TO_LDS_B128 (ASYNCcnt),
// overlapping the next tile's global->LDS copy with WMMA on the current one.
// Softmax weight computed exp/log-free: exp(-acosh(c)) == 1/(c + sqrt(c^2-1));
// scores <= 0 so no running-max rescaling is needed (p in (0,1]).
// ---------------------------------------------------------------------------

typedef __attribute__((ext_vector_type(16))) _Float16 v16h;
typedef __attribute__((ext_vector_type(8)))  float    v8f;
typedef unsigned int u32;
typedef __attribute__((ext_vector_type(4)))  u32      u32x4;

#define NSEQ   2048
#define DMODEL 1024
#define NHEAD  16
#define HDIM   64
#define LSTR   72   // LDS row stride in halves (144 B: 16B-aligned, conflict pad)

// Per-lane async copy of 16 bytes global -> LDS (tracked by ASYNCcnt).
// LDS aperture: flat addr[31:0] == LDS byte offset (ISA 10.2).
__device__ __forceinline__ void async_cp16(void* lds, const void* gp) {
  unsigned dst = (unsigned)(unsigned long long)lds;
  asm volatile("global_load_async_to_lds_b128 %0, %1, off"
               :: "v"(dst), "v"((unsigned long long)gp)
               : "memory");
}
__device__ __forceinline__ void wait_async0() {
  asm volatile("s_wait_asynccnt 0" ::: "memory");
}

// Build a v16h fragment from two 16-byte chunks (halves 0..7 and 8..15).
__device__ __forceinline__ v16h frag2(const _Float16* p0, const _Float16* p1) {
  union { u32x4 q[2]; v16h h; } u;
  u.q[0] = *(const u32x4*)p0;
  u.q[1] = *(const u32x4*)p1;
  return u.h;
}
// A-fragment (16x32 f16): runs at k = 8*half and k = 16 + 8*half.
__device__ __forceinline__ v16h frag_a(const _Float16* row, int half) {
  return frag2(row + 8 * half, row + 16 + 8 * half);
}
// B-fragment (32x16 f16): runs at k = 8*half and k = 16 + 8*half.
__device__ __forceinline__ v16h frag_b(const _Float16* colbase, int half) {
  return frag2(colbase + 8 * half, colbase + 16 + 8 * half);
}

// butterfly sum over the 16-lane column group of a WMMA C tile
__device__ __forceinline__ float grp16_sum(float v) {
  v += __shfl_xor(v, 1, 32);
  v += __shfl_xor(v, 2, 32);
  v += __shfl_xor(v, 4, 32);
  v += __shfl_xor(v, 8, 32);
  return v;
}

// ---------------------------------------------------------------------------
__global__ void cvt_f32_f16(const float* __restrict__ src,
                            _Float16* __restrict__ dst, int n) {
  for (int i = blockIdx.x * blockDim.x + threadIdx.x; i < n;
       i += gridDim.x * blockDim.x)
    dst[i] = (_Float16)src[i];
}

__global__ void zero_f32(float* __restrict__ p, int n) {
  for (int i = blockIdx.x * blockDim.x + threadIdx.x; i < n;
       i += gridDim.x * blockDim.x)
    p[i] = 0.0f;
}

// ---------------------------------------------------------------------------
// Y[n,j] = sum_d A[n,d] * W[j,d]   (A:[2048,1024] f16, W:[1024,1024] f16)
// 256 threads = 8 waves; workgroup tile 128(M) x 64(N); wave tile 32x32.
// K staged 64 wide, double-buffered via async LDS loads.
// NORMS: atomicAdd per-(head,row) sum of squares of Y (head = 64-col block).
// OUT32: write f32 (final projection). TRANS: write Y^T f16 [D, N] (for V).
// ---------------------------------------------------------------------------
template <bool NORMS, bool OUT32, bool TRANS>
__global__ __launch_bounds__(256) void wmma_gemm(
    const _Float16* __restrict__ A, const _Float16* __restrict__ W,
    _Float16* __restrict__ Yh, float* __restrict__ Yf,
    float* __restrict__ norm) {
  __shared__ alignas(16) _Float16 As[2][128 * LSTR];
  __shared__ alignas(16) _Float16 Bs[2][64 * LSTR];

  const int tid  = threadIdx.x;
  const int lane = tid & 31;
  const int wid  = tid >> 5;   // 0..7
  const int wm   = wid >> 1;   // 0..3 (M)
  const int wn   = wid & 1;    // 0..1 (N)
  const int col  = lane & 15;
  const int half = lane >> 4;
  const int mBase = blockIdx.x * 128;
  const int nBase = blockIdx.y * 64;

  auto issue = [&](int kb, int buf) {
#pragma unroll
    for (int i = 0; i < 4; ++i) {   // A tile 128x64: 1024 chunks, 4/thread
      int c = tid + 256 * i, row = c >> 3, off = (c & 7) * 8;
      async_cp16(&As[buf][row * LSTR + off],
                 &A[(mBase + row) * DMODEL + kb + off]);
    }
#pragma unroll
    for (int i = 0; i < 2; ++i) {   // B tile 64x64: 512 chunks, 2/thread
      int c = tid + 256 * i, row = c >> 3, off = (c & 7) * 8;
      async_cp16(&Bs[buf][row * LSTR + off],
                 &W[(nBase + row) * DMODEL + kb + off]);
    }
  };

  v8f zero = {};
  v8f acc[2][2];
  acc[0][0] = zero; acc[0][1] = zero; acc[1][0] = zero; acc[1][1] = zero;

  issue(0, 0);
  for (int kb = 0; kb < DMODEL; kb += 64) {
    const int cur = (kb >> 6) & 1;
    wait_async0();     // my async copies (into cur) are done
    __syncthreads();   // => everyone's copies are done & visible
    if (kb + 64 < DMODEL) issue(kb + 64, cur ^ 1);

#pragma unroll
    for (int kk = 0; kk < 2; ++kk) {
      v16h a[2], b[2];
#pragma unroll
      for (int i = 0; i < 2; ++i)
        a[i] = frag_a(&As[cur][(wm * 32 + i * 16 + col) * LSTR + kk * 32], half);
#pragma unroll
      for (int j = 0; j < 2; ++j)
        b[j] = frag_b(&Bs[cur][(wn * 32 + j * 16 + col) * LSTR + kk * 32], half);
#pragma unroll
      for (int i = 0; i < 2; ++i)
#pragma unroll
        for (int j = 0; j < 2; ++j)
          acc[i][j] = __builtin_amdgcn_wmma_f32_16x16x32_f16(
              false, a[i], false, b[j], (short)0, acc[i][j], false, false);
    }
  }

  // epilogue: store + optional per-head row norms from f32 accumulators
#pragma unroll
  for (int i = 0; i < 2; ++i) {
#pragma unroll
    for (int j = 0; j < 2; ++j) {
      int gcol = nBase + wn * 32 + j * 16 + col;
#pragma unroll
      for (int r = 0; r < 8; ++r) {
        int grow = mBase + wm * 32 + i * 16 + r + 8 * half;
        float v  = acc[i][j][r];
        if constexpr (OUT32)      Yf[grow * DMODEL + gcol] = v;
        else if constexpr (TRANS) Yh[gcol * NSEQ + grow]   = (_Float16)v;
        else                      Yh[grow * DMODEL + gcol] = (_Float16)v;
      }
    }
    if constexpr (NORMS) {
      int head = nBase >> 6;  // 64-wide workgroup N-tile == one head
#pragma unroll
      for (int r = 0; r < 8; ++r) {
        float v = acc[i][0][r] * acc[i][0][r] + acc[i][1][r] * acc[i][1][r];
        v = grp16_sum(v);  // this wave's 32 columns; x2 N-waves via atomic
        if (col == 0) {
          int grow = mBase + wm * 32 + i * 16 + r + 8 * half;
          atomicAdd(&norm[head * NSEQ + grow], v);
        }
      }
    }
  }
}

// ---------------------------------------------------------------------------
// Flash-style hyperbolic attention: one workgroup = 64 query rows x 1 head.
// 128 threads = 4 waves; each wave owns 16 query rows, full 64-wide output.
// K tile staged row-major [key][hd]; V staged from V^T as [hd][key] so the
// P.V B-fragments are contiguous 16-byte LDS loads. Double-buffered async
// staging overlaps the next key tile's copy with compute on the current one.
// ---------------------------------------------------------------------------
__global__ __launch_bounds__(128) void hyper_attn(
    const _Float16* __restrict__ Qh, const _Float16* __restrict__ Kh,
    const _Float16* __restrict__ Vt, const float* __restrict__ qnorm,
    const float* __restrict__ knorm, _Float16* __restrict__ Oh) {
  __shared__ alignas(16) _Float16 Kl[2][64 * LSTR];    // [key][hd]
  __shared__ alignas(16) _Float16 Vl[2][64 * LSTR];    // [hd][key] (from V^T)
  __shared__ alignas(16) _Float16 Pl[4 * 16 * LSTR];   // per-wave P staging

  const int tid  = threadIdx.x;
  const int lane = tid & 31;
  const int w    = tid >> 5;  // 0..3
  const int col  = lane & 15;
  const int half = lane >> 4;
  const int qBase = blockIdx.x * 64;
  const int h     = blockIdx.y;
  const int hOff  = h * HDIM;

  auto issue = [&](int kt, int buf) {
#pragma unroll
    for (int i = 0; i < 4; ++i) {   // 512 chunks per tile, 4/thread each
      int c = tid + 128 * i, row = c >> 3, off = (c & 7) * 8;
      async_cp16(&Kl[buf][row * LSTR + off],
                 &Kh[(kt + row) * DMODEL + hOff + off]);
      async_cp16(&Vl[buf][row * LSTR + off],
                 &Vt[(hOff + row) * NSEQ + kt + off]);
    }
  };

  issue(0, 0);

  // preload Q fragments (16 rows x 64 = two K-chunks of 32): 16B global loads
  v16h aq[2];
  {
    const _Float16* qp = Qh + (qBase + w * 16 + col) * DMODEL + hOff;
    aq[0] = frag_a(qp, half);
    aq[1] = frag_a(qp + 32, half);
  }
  float qn[8], qn1[8];
#pragma unroll
  for (int r = 0; r < 8; ++r) {
    qn[r]  = qnorm[h * NSEQ + qBase + w * 16 + r + 8 * half];
    qn1[r] = 1.0f - qn[r];
  }

  float lrow[8];
  v8f zero = {};
  v8f o[4];
#pragma unroll
  for (int r = 0; r < 8; ++r) lrow[r] = 0.0f;
#pragma unroll
  for (int j = 0; j < 4; ++j) o[j] = zero;

  for (int kt = 0; kt < NSEQ; kt += 64) {
    const int cur = (kt >> 6) & 1;
    wait_async0();     // my async copies (into cur) complete
    __syncthreads();   // => whole tile visible; prev buffer free to overwrite
    if (kt + 64 < NSEQ) issue(kt + 64, cur ^ 1);

    // S = Q K^T  (B[k][n'] = K[n'][k]: contiguous along k in the K tile)
    v8f s[4];
#pragma unroll
    for (int j = 0; j < 4; ++j) {
      v8f z = zero;
#pragma unroll
      for (int kk = 0; kk < 2; ++kk) {
        v16h b = frag_b(&Kl[cur][(j * 16 + col) * LSTR + kk * 32], half);
        z = __builtin_amdgcn_wmma_f32_16x16x32_f16(false, aq[kk], false, b,
                                                   (short)0, z, false, false);
      }
      s[j] = z;
    }

    // p = exp(-acosh(c)) = 1/(c + sqrt(c^2-1)),  c = 1 + z,  z >= 0
    float kn[4], kn1[4];
#pragma unroll
    for (int j = 0; j < 4; ++j) {
      kn[j]  = knorm[h * NSEQ + kt + j * 16 + col];
      kn1[j] = 1.0f - kn[j];
    }
#pragma unroll
    for (int j = 0; j < 4; ++j)
#pragma unroll
      for (int r = 0; r < 8; ++r) {
        float qk    = s[j][r];
        float diff  = fmaxf(qn[r] + kn[j] - 2.0f * qk, 0.0f);
        float denom = fmaxf(qn1[r] * kn1[j], 1e-6f);
        float zz    = 2.0f * diff * __builtin_amdgcn_rcpf(denom);
        float root  = __builtin_amdgcn_sqrtf(zz * (zz + 2.0f));
        float p     = __builtin_amdgcn_rcpf(1.0f + zz + root);
        s[j][r] = p;
        lrow[r] += p;
      }

    // stage P (f16) in per-wave LDS region, re-read as WMMA A fragments
    // (LDS ops are hardware in-order within a wave: no barrier needed)
#pragma unroll
    for (int j = 0; j < 4; ++j)
#pragma unroll
      for (int r = 0; r < 8; ++r)
        Pl[(w * 16 + r + 8 * half) * LSTR + j * 16 + col] = (_Float16)s[j][r];

    // O += P V   (B[k][n'] = V[k][n'] = Vt[n'][k]: contiguous in the V tile)
#pragma unroll
    for (int kk = 0; kk < 2; ++kk) {
      v16h pa = frag_a(&Pl[(w * 16 + col) * LSTR + kk * 32], half);
#pragma unroll
      for (int j = 0; j < 4; ++j) {
        v16h vb = frag_b(&Vl[cur][(j * 16 + col) * LSTR + kk * 32], half);
        o[j] = __builtin_amdgcn_wmma_f32_16x16x32_f16(false, pa, false, vb,
                                                      (short)0, o[j], false,
                                                      false);
      }
    }
  }

  // single end-of-loop reduction of the row sums, then normalize + store
  float rinv[8];
#pragma unroll
  for (int r = 0; r < 8; ++r)
    rinv[r] = __builtin_amdgcn_rcpf(grp16_sum(lrow[r]));
#pragma unroll
  for (int j = 0; j < 4; ++j)
#pragma unroll
    for (int r = 0; r < 8; ++r) {
      int grow = qBase + w * 16 + r + 8 * half;
      Oh[grow * DMODEL + hOff + j * 16 + col] = (_Float16)(o[j][r] * rinv[r]);
    }
}

// ---------------------------------------------------------------------------
extern "C" void kernel_launch(void* const* d_in, const int* in_sizes, int n_in,
                              void* d_out, int out_size, void* d_ws,
                              size_t ws_size, hipStream_t stream) {
  (void)in_sizes; (void)n_in; (void)out_size; (void)ws_size;
  const float* x  = (const float*)d_in[0];
  const float* Wq = (const float*)d_in[1];
  const float* Wk = (const float*)d_in[2];
  const float* Wv = (const float*)d_in[3];
  const float* Wo = (const float*)d_in[4];

  char* ws = (char*)d_ws;
  const size_t MB = 1024 * 1024;
  _Float16* xh  = (_Float16*)(ws + 0 * MB);   // 4 MB
  _Float16* wqh = (_Float16*)(ws + 4 * MB);   // 2 MB each
  _Float16* wkh = (_Float16*)(ws + 6 * MB);
  _Float16* wvh = (_Float16*)(ws + 8 * MB);
  _Float16* woh = (_Float16*)(ws + 10 * MB);
  _Float16* Qh  = (_Float16*)(ws + 12 * MB);  // 4 MB each
  _Float16* Kh  = (_Float16*)(ws + 16 * MB);
  _Float16* Vt  = (_Float16*)(ws + 20 * MB);  // V^T  [D, N]
  _Float16* Oh  = (_Float16*)(ws + 24 * MB);
  float*    qn  = (float*)(ws + 28 * MB);     // 128 KB
  float*    kn  = (float*)(ws + 28 * MB + 128 * 1024);

  cvt_f32_f16<<<1024, 256, 0, stream>>>(x, xh, NSEQ * DMODEL);
  cvt_f32_f16<<<512, 256, 0, stream>>>(Wq, wqh, DMODEL * DMODEL);
  cvt_f32_f16<<<512, 256, 0, stream>>>(Wk, wkh, DMODEL * DMODEL);
  cvt_f32_f16<<<512, 256, 0, stream>>>(Wv, wvh, DMODEL * DMODEL);
  cvt_f32_f16<<<512, 256, 0, stream>>>(Wo, woh, DMODEL * DMODEL);
  zero_f32<<<64, 256, 0, stream>>>(qn, 2 * NHEAD * NSEQ);  // qn+kn contiguous

  dim3 gg(NSEQ / 128, DMODEL / 64);
  wmma_gemm<true,  false, false><<<gg, 256, 0, stream>>>(xh, wqh, Qh, nullptr, qn);
  wmma_gemm<true,  false, false><<<gg, 256, 0, stream>>>(xh, wkh, Kh, nullptr, kn);
  wmma_gemm<false, false, true ><<<gg, 256, 0, stream>>>(xh, wvh, Vt, nullptr, nullptr);

  hyper_attn<<<dim3(NSEQ / 64, NHEAD), 128, 0, stream>>>(Qh, Kh, Vt, qn, kn, Oh);

  wmma_gemm<false, true, false><<<gg, 256, 0, stream>>>(Oh, woh, nullptr,
                                                        (float*)d_out, nullptr);
}
